// ResidualSSM_33225867002394
// MI455X (gfx1250) — matrix operand order repaired and checked by reading, not compiled, not run on hardware
//
#include <hip/hip_runtime.h>
#include <hip/hip_bf16.h>

// ---------------------------------------------------------------------------
// CDNA5 (gfx1250) implementation of the bidirectional selective-SSM model.
// wave32, WMMA bf16 16x16x32 f32-accumulate, double-buffered LDS staging,
// GLOBAL_LOAD_ASYNC_TO_LDS_B128 for bf16 tiles (ASYNCcnt-tracked).
// ---------------------------------------------------------------------------

typedef __bf16 v4bf  __attribute__((ext_vector_type(4)));
typedef __bf16 v8bf  __attribute__((ext_vector_type(8)));
typedef __bf16 v16bf __attribute__((ext_vector_type(16)));
typedef float  v8f   __attribute__((ext_vector_type(8)));
typedef int    v4i   __attribute__((ext_vector_type(4)));

#define B_SZ   4096
#define T_SZ   12
#define M_TOT  (B_SZ * T_SZ)   // 49152
#define K_TOT  1770
#define K_PAD  1792
#define NPROJ  160              // [Wp_f | Wp_b] columns
#define NC     234
#define NC_PAD 240

#if defined(__HIP_DEVICE_COMPILE__) && __has_builtin(__builtin_amdgcn_global_load_async_to_lds_b128)
#define HAS_ASYNC_LDS 1
#else
#define HAS_ASYNC_LDS 0
#endif

#if HAS_ASYNC_LDS
typedef __attribute__((address_space(3))) v4i lds_v4i;
__device__ __forceinline__ void async_b128(const void* g, void* l) {
  __builtin_amdgcn_global_load_async_to_lds_b128((v4i*)g, (lds_v4i*)l, 0, 0);
}
__device__ __forceinline__ void wait_async0() {
#if __has_builtin(__builtin_amdgcn_s_wait_asynccnt)
  __builtin_amdgcn_s_wait_asynccnt(0);
#else
  asm volatile("s_wait_asynccnt 0x0" ::: "memory");
#endif
}
#else
__device__ __forceinline__ void async_b128(const void* g, void* l) {
  *(uint4*)l = *(const uint4*)g;  // fallback: VGPR round-trip
}
__device__ __forceinline__ void wait_async0() {}
#endif

// Build the 16-element (per-lane) WMMA operand from two contiguous 8x bf16
// LDS chunks (matches both the A 16x32 and B 32x16 VGPR layouts).
__device__ __forceinline__ v16bf mk16(const __bf16* lo, const __bf16* hi) {
  v16bf r;
  v8bf a = *(const v8bf*)lo;
  v8bf b = *(const v8bf*)hi;
#pragma unroll
  for (int i = 0; i < 8; ++i) { r[i] = a[i]; r[8 + i] = b[i]; }
  return r;
}

__device__ __forceinline__ v8f wmma_bf16(v16bf a, v16bf b, v8f c) {
  return __builtin_amdgcn_wmma_f32_16x16x32_bf16(false, a, false, b,
                                                 (short)0, c, false, false);
}

// ---------------------------------------------------------------------------
// Prep: convert/transpose weights to bf16 once (all weights are L2-resident).
// ---------------------------------------------------------------------------
__global__ __launch_bounds__(256) void k_prep(
    const float* __restrict__ W_in, const float* __restrict__ Wp_f,
    const float* __restrict__ Wp_b, const float* __restrict__ W_merge,
    const float* __restrict__ W_out, const float* __restrict__ bp_f,
    const float* __restrict__ bp_b,
    __bf16* __restrict__ WtG, __bf16* __restrict__ WptG,
    __bf16* __restrict__ WmtG, __bf16* __restrict__ WotG,
    float* __restrict__ bpcat) {
  int idx = blockIdx.x * 256 + threadIdx.x;
  if (idx < 64 * K_PAD) {
    int n = idx / K_PAD, k = idx % K_PAD;
    WtG[idx] = (__bf16)((k < K_TOT) ? W_in[(size_t)k * 64 + n] : 0.0f);
    return;
  }
  idx -= 64 * K_PAD;
  if (idx < NPROJ * 64) {
    int n = idx / 64, k = idx % 64;
    float v = (n < 80) ? Wp_f[k * 80 + n] : Wp_b[k * 80 + (n - 80)];
    WptG[idx] = (__bf16)v;
    return;
  }
  idx -= NPROJ * 64;
  if (idx < 64 * 128) {
    int n = idx / 128, k = idx % 128;
    WmtG[idx] = (__bf16)W_merge[k * 64 + n];
    return;
  }
  idx -= 64 * 128;
  if (idx < NC_PAD * 64) {
    int n = idx / 64, k = idx % 64;
    WotG[idx] = (__bf16)((n < NC) ? W_out[k * NC + n] : 0.0f);
    return;
  }
  idx -= NC_PAD * 64;
  if (idx < NPROJ) bpcat[idx] = (idx < 80) ? bp_f[idx] : bp_b[idx - 80];
}

// ---------------------------------------------------------------------------
// meta[b][c] = concat(site_e, hour_e) @ W_meta + b_meta   (K=16, trivial)
// ---------------------------------------------------------------------------
__global__ __launch_bounds__(256) void k_meta(
    const int* __restrict__ site_ids, const int* __restrict__ hours,
    const float* __restrict__ site_emb, const float* __restrict__ hour_emb,
    const float* __restrict__ W_meta, const float* __restrict__ b_meta,
    float* __restrict__ meta) {
  int idx = blockIdx.x * 256 + threadIdx.x;
  if (idx >= B_SZ * 64) return;
  int b = idx >> 6, c = idx & 63;
  int s = min(max(site_ids[b], 0), 19);
  int h = min(max(hours[b], 0), 23);
  float acc = b_meta[c];
#pragma unroll
  for (int j = 0; j < 8; ++j) acc += site_emb[s * 8 + j] * W_meta[j * 64 + c];
#pragma unroll
  for (int j = 0; j < 8; ++j) acc += hour_emb[h * 8 + j] * W_meta[(8 + j) * 64 + c];
  meta[idx] = acc;
}

// ---------------------------------------------------------------------------
// Stage 1 (HBM bound, ~356MB in): h = gelu(LN(x @ W_in + b_in)) + meta + pos
// 32x64 tile/block, 8 waves = (m_half 0..1) x (n_quad 0..3).
// Double-buffered LDS; weights staged via async-to-LDS.
// ---------------------------------------------------------------------------
__global__ __launch_bounds__(256) void k_in(
    const float* __restrict__ emb, const float* __restrict__ scores,
    const float* __restrict__ b_in, const float* __restrict__ ln_g,
    const float* __restrict__ ln_b, const __bf16* __restrict__ WtG,
    const float* __restrict__ meta, const float* __restrict__ pos,
    float* __restrict__ h32, __bf16* __restrict__ hbf) {
  __shared__ __bf16 As[2][32][72];
  __shared__ __bf16 Ws[2][64][72];
  __shared__ float out1[32][64];
  __shared__ float muA[32], rsA[32];

  const int tid = threadIdx.x;
  const int lane = tid & 31;
  const int wave = tid >> 5;
  const int m_half = wave >> 2;
  const int n_quad = wave & 3;
  const int half = lane >> 4;
  const int l15 = lane & 15;
  const size_t m_base = (size_t)blockIdx.x * 32;

  const int ar = tid >> 4;             // A-loader row (0..15)
  const int ac4 = (tid & 15) * 4;      // A-loader col (step 4)
  const int wn = tid >> 2;             // W-loader row (0..63)
  const int wk = (tid & 3) * 16;       // W-loader col (step 16)

  auto load_A = [&](int buf, int ch) {
    const int k0 = ch * 64;
#pragma unroll
    for (int p = 0; p < 2; ++p) {
      const int row = ar + p * 16;
      float v0, v1, v2, v3;
      if (k0 < 1536) {
        const float4 v =
            *(const float4*)(emb + (m_base + row) * 1536 + k0 + ac4);
        v0 = v.x; v1 = v.y; v2 = v.z; v3 = v.w;
        if (ch + 1 < 24)  // prefetch next emb chunk
          __builtin_prefetch(emb + (m_base + row) * 1536 + k0 + 64 + ac4, 0, 1);
      } else {
        const float* srow = scores + (m_base + row) * 234;
        const int cb = k0 - 1536 + ac4;
        v0 = (cb + 0 < 234) ? srow[cb + 0] : 0.f;
        v1 = (cb + 1 < 234) ? srow[cb + 1] : 0.f;
        v2 = (cb + 2 < 234) ? srow[cb + 2] : 0.f;
        v3 = (cb + 3 < 234) ? srow[cb + 3] : 0.f;
      }
      v4bf pk = {(__bf16)v0, (__bf16)v1, (__bf16)v2, (__bf16)v3};
      *(v4bf*)&As[buf][row][ac4] = pk;
    }
  };
  auto load_W = [&](int buf, int ch) {
    const int k0 = ch * 64;
    async_b128(WtG + (size_t)wn * K_PAD + k0 + wk, &Ws[buf][wn][wk]);
    async_b128(WtG + (size_t)wn * K_PAD + k0 + wk + 8, &Ws[buf][wn][wk + 8]);
  };

  v8f acc = {0.f, 0.f, 0.f, 0.f, 0.f, 0.f, 0.f, 0.f};

  load_A(0, 0);
  load_W(0, 0);
  wait_async0();
  __syncthreads();

  for (int ch = 0; ch < 28; ++ch) {
    const int cur = ch & 1, nxt = cur ^ 1;
    if (ch + 1 < 28) {  // stage next chunk while computing this one
      load_A(nxt, ch + 1);
      load_W(nxt, ch + 1);
    }
    const __bf16* arow = &As[cur][m_half * 16 + l15][0];
    const __bf16* brow = &Ws[cur][n_quad * 16 + l15][0];
#pragma unroll
    for (int kc = 0; kc < 64; kc += 32) {
      v16bf a = mk16(arow + kc + half * 8, arow + kc + 16 + half * 8);
      v16bf b = mk16(brow + kc + half * 8, brow + kc + 16 + half * 8);
      acc = wmma_bf16(a, b, acc);
    }
    wait_async0();
    __syncthreads();
  }

  // ---- epilogue: C layout -> LDS, LN + GELU + meta + pos ----
#pragma unroll
  for (int r = 0; r < 8; ++r)
    out1[m_half * 16 + half * 8 + r][n_quad * 16 + l15] = acc[r];
  __syncthreads();
  if (tid < 32) {
    float s = 0.f, s2 = 0.f;
    for (int c = 0; c < 64; ++c) {
      float v = out1[tid][c] + b_in[c];
      s += v; s2 += v * v;
    }
    float mu = s * (1.f / 64.f);
    float var = s2 * (1.f / 64.f) - mu * mu;
    muA[tid] = mu;
    rsA[tid] = rsqrtf(var + 1e-5f);
  }
  __syncthreads();
  {
    const int row = tid >> 3, c0 = (tid & 7) * 8;
    const size_t m = m_base + row;
    const int bidx = (int)(m / T_SZ), t = (int)(m % T_SZ);
    const float mu = muA[row], rs = rsA[row];
#pragma unroll
    for (int j = 0; j < 8; ++j) {
      const int c = c0 + j;
      float v = out1[row][c] + b_in[c];
      float y = (v - mu) * rs * ln_g[c] + ln_b[c];
      float ge = 0.5f * y * (1.f + erff(y * 0.70710678118654752f));
      float hv = ge + meta[(size_t)bidx * 64 + c] + pos[t * 64 + c];
      h32[m * 64 + c] = hv;
      hbf[m * 64 + c] = (__bf16)hv;
    }
  }
}

// ---------------------------------------------------------------------------
// Stage 2: proj = h @ [Wp_f|Wp_b] + [bp_f|bp_b]   (M=49152, K=64, N=160)
// A and W staged via async-to-LDS (both bf16 in global).
// ---------------------------------------------------------------------------
__global__ __launch_bounds__(256) void k_proj(
    const __bf16* __restrict__ hbf, const __bf16* __restrict__ WptG,
    const float* __restrict__ bpcat, float* __restrict__ proj) {
  __shared__ __bf16 As[32][72];
  __shared__ __bf16 Ws[NPROJ][72];

  const int tid = threadIdx.x;
  const int lane = tid & 31;
  const int wave = tid >> 5;
  const int m_half = wave >> 2;
  const int n_quad = wave & 3;
  const int half = lane >> 4;
  const int l15 = lane & 15;
  const size_t m_base = (size_t)blockIdx.x * 32;

  {  // A tile: 32x64 bf16 = one b128 per thread
    const int row = tid >> 3, c8 = (tid & 7) * 8;
    async_b128(hbf + (m_base + row) * 64 + c8, &As[row][c8]);
  }
  for (int i = tid; i < NPROJ * 8; i += 256) {  // whole weight: 160x64 bf16
    const int row = i >> 3, k8 = (i & 7) * 8;
    async_b128(WptG + row * 64 + k8, &Ws[row][k8]);
  }
  wait_async0();
  __syncthreads();

  const __bf16* arow = &As[m_half * 16 + l15][0];
  for (int nt = n_quad; nt < NPROJ / 16; nt += 4) {
    v8f acc = {0.f, 0.f, 0.f, 0.f, 0.f, 0.f, 0.f, 0.f};
    const __bf16* brow = &Ws[nt * 16 + l15][0];
#pragma unroll
    for (int kc = 0; kc < 64; kc += 32) {
      v16bf a = mk16(arow + kc + half * 8, arow + kc + 16 + half * 8);
      v16bf b = mk16(brow + kc + half * 8, brow + kc + 16 + half * 8);
      acc = wmma_bf16(a, b, acc);
    }
    const int n = nt * 16 + l15;
    const float bb = bpcat[n];
#pragma unroll
    for (int r = 0; r < 8; ++r) {
      const size_t m = m_base + m_half * 16 + half * 8 + r;
      proj[m * NPROJ + n] = acc[r] + bb;
    }
  }
}

// ---------------------------------------------------------------------------
// Stage 3: selective-SSM scan, one thread per (b, dir, d). State (8) in regs.
// Backward direction scans t descending (== scanning the reversed sequence)
// and stores at the reversed position, so no explicit flips are needed.
// ---------------------------------------------------------------------------
__global__ __launch_bounds__(128) void k_scan(
    const float* __restrict__ proj, const float* __restrict__ h32,
    const float* __restrict__ AlogF, const float* __restrict__ AlogB,
    const float* __restrict__ Df, const float* __restrict__ Db,
    __bf16* __restrict__ hcat) {
  const int b = blockIdx.x;
  const int tid = threadIdx.x;
  const int dir = tid >> 6;
  const int d = tid & 63;
  const float* Alog = dir ? AlogB : AlogF;
  const float Dv = dir ? Db[d] : Df[d];
  float A[8];
#pragma unroll
  for (int n = 0; n < 8; ++n) A[n] = -__expf(Alog[d * 8 + n]);
  float st[8] = {0.f, 0.f, 0.f, 0.f, 0.f, 0.f, 0.f, 0.f};
#pragma unroll
  for (int t = 0; t < T_SZ; ++t) {
    const int tt = dir ? (T_SZ - 1 - t) : t;
    const size_t m = (size_t)b * T_SZ + tt;
    const float* pr = proj + m * NPROJ + dir * 80;
    const float x = h32[m * 64 + d];
    const float pd = pr[d];
    const float delta = (pd > 20.f) ? pd : log1pf(__expf(pd));
    const float dx = delta * x;
    float y = 0.f;
#pragma unroll
    for (int n = 0; n < 8; ++n) {
      const float s = __expf(delta * A[n]) * st[n] + dx * pr[64 + n];
      st[n] = s;
      y += s * pr[72 + n];
    }
    hcat[m * 128 + dir * 64 + d] = (__bf16)(y + x * Dv);
  }
}

// ---------------------------------------------------------------------------
// Stage 4: out = LN(hcat @ W_merge + b_merge + residual) @ W_out + b_out
// Two chained WMMA GEMMs; W_out tile reuses merge-stage LDS (async staged).
// ---------------------------------------------------------------------------
__global__ __launch_bounds__(256) void k_out(
    const __bf16* __restrict__ hcat, const float* __restrict__ h32,
    const __bf16* __restrict__ WmtG, const __bf16* __restrict__ WotG,
    const float* __restrict__ b_merge, const float* __restrict__ lng,
    const float* __restrict__ lnb, const float* __restrict__ b_out,
    float* __restrict__ out) {
  __shared__ __attribute__((aligned(16))) char smem[NC_PAD * 72 * 2];  // 34560B
  __shared__ float out1[32][64];
  __shared__ float muA[32], rsA[32];
  __shared__ __bf16 A2s[32][72];

  __bf16* A1s = (__bf16*)smem;                   // [32][136]
  __bf16* Wms = (__bf16*)(smem + 32 * 136 * 2);  // [64][136]
  __bf16* Wos = (__bf16*)smem;                   // [240][72] (phase 2)

  const int tid = threadIdx.x;
  const int lane = tid & 31;
  const int wave = tid >> 5;
  const int m_half = wave >> 2;
  const int n_quad = wave & 3;
  const int half = lane >> 4;
  const int l15 = lane & 15;
  const size_t m_base = (size_t)blockIdx.x * 32;

  {  // A1: 32x128 bf16
    const int row = tid >> 3, cc = (tid & 7) * 16;
    async_b128(hcat + (m_base + row) * 128 + cc, &A1s[row * 136 + cc]);
    async_b128(hcat + (m_base + row) * 128 + cc + 8, &A1s[row * 136 + cc + 8]);
  }
  for (int i = tid; i < 64 * 16; i += 256) {  // Wm: 64x128 bf16
    const int row = i >> 4, k8 = (i & 15) * 8;
    async_b128(WmtG + row * 128 + k8, &Wms[row * 136 + k8]);
  }
  wait_async0();
  __syncthreads();

  // GEMM1: (32x128) @ (128x64)
  v8f acc = {0.f, 0.f, 0.f, 0.f, 0.f, 0.f, 0.f, 0.f};
  {
    const __bf16* arow = &A1s[(m_half * 16 + l15) * 136];
    const __bf16* brow = &Wms[(n_quad * 16 + l15) * 136];
#pragma unroll
    for (int kc = 0; kc < 128; kc += 32) {
      v16bf a = mk16(arow + kc + half * 8, arow + kc + 16 + half * 8);
      v16bf b = mk16(brow + kc + half * 8, brow + kc + 16 + half * 8);
      acc = wmma_bf16(a, b, acc);
    }
  }
#pragma unroll
  for (int r = 0; r < 8; ++r) {
    const int mm = m_half * 16 + half * 8 + r;
    const int nn = n_quad * 16 + l15;
    out1[mm][nn] = acc[r] + b_merge[nn] + h32[(m_base + mm) * 64 + nn];
  }
  __syncthreads();  // all GEMM1 LDS reads retired -> smem reusable

  // Stage W_out tile over the merge-stage LDS region.
  for (int i = tid; i < NC_PAD * 8; i += 256) {
    const int row = i >> 3, k8 = (i & 7) * 8;
    async_b128(WotG + row * 64 + k8, &Wos[row * 72 + k8]);
  }
  if (tid < 32) {
    float s = 0.f, s2 = 0.f;
    for (int c = 0; c < 64; ++c) {
      float v = out1[tid][c];
      s += v; s2 += v * v;
    }
    float mu = s * (1.f / 64.f);
    float var = s2 * (1.f / 64.f) - mu * mu;
    muA[tid] = mu;
    rsA[tid] = rsqrtf(var + 1e-5f);
  }
  wait_async0();
  __syncthreads();
  {
    const int row = tid >> 3, c0 = (tid & 7) * 8;
    const float mu = muA[row], rs = rsA[row];
#pragma unroll
    for (int j = 0; j < 8; ++j) {
      const int c = c0 + j;
      float y = (out1[row][c] - mu) * rs * lng[c] + lnb[c];
      A2s[row][c] = (__bf16)y;
    }
  }
  __syncthreads();

  // GEMM2: (32x64) @ (64x240), store masked to NC=234.
  const __bf16* a2row = &A2s[m_half * 16 + l15][0];
  for (int nt = n_quad; nt < NC_PAD / 16; nt += 4) {
    v8f c2 = {0.f, 0.f, 0.f, 0.f, 0.f, 0.f, 0.f, 0.f};
    const __bf16* brow = &Wos[(nt * 16 + l15) * 72];
#pragma unroll
    for (int kc = 0; kc < 64; kc += 32) {
      v16bf a = mk16(a2row + kc + half * 8, a2row + kc + 16 + half * 8);
      v16bf b = mk16(brow + kc + half * 8, brow + kc + 16 + half * 8);
      c2 = wmma_bf16(a, b, c2);
    }
    const int n = nt * 16 + l15;
    if (n < NC) {
      const float bo = b_out[n];
#pragma unroll
      for (int r = 0; r < 8; ++r) {
        const size_t m = m_base + m_half * 16 + half * 8 + r;
        out[m * NC + n] = c2[r] + bo;
      }
    }
  }
}

// ---------------------------------------------------------------------------
extern "C" void kernel_launch(void* const* d_in, const int* in_sizes, int n_in,
                              void* d_out, int out_size, void* d_ws,
                              size_t ws_size, hipStream_t stream) {
  (void)in_sizes; (void)n_in; (void)out_size; (void)ws_size;
  const float* emb      = (const float*)d_in[0];
  const float* scores   = (const float*)d_in[1];
  const int*   site_ids = (const int*)d_in[2];
  const int*   hours    = (const int*)d_in[3];
  const float* W_in     = (const float*)d_in[4];
  const float* b_in     = (const float*)d_in[5];
  const float* ln_in_g  = (const float*)d_in[6];
  const float* ln_in_b  = (const float*)d_in[7];
  const float* site_emb = (const float*)d_in[8];
  const float* hour_emb = (const float*)d_in[9];
  const float* W_meta   = (const float*)d_in[10];
  const float* b_meta   = (const float*)d_in[11];
  const float* pos_enc  = (const float*)d_in[12];
  const float* A_log_f  = (const float*)d_in[13];
  const float* Wp_f     = (const float*)d_in[14];
  const float* bp_f     = (const float*)d_in[15];
  const float* D_f      = (const float*)d_in[16];
  const float* A_log_b  = (const float*)d_in[17];
  const float* Wp_b     = (const float*)d_in[18];
  const float* bp_b     = (const float*)d_in[19];
  const float* D_b      = (const float*)d_in[20];
  const float* W_merge  = (const float*)d_in[21];
  const float* b_merge  = (const float*)d_in[22];
  const float* ln_g     = (const float*)d_in[23];
  const float* ln_b     = (const float*)d_in[24];
  const float* W_out    = (const float*)d_in[25];
  const float* b_out    = (const float*)d_in[26];

  char* ws = (char*)d_ws;
  float*  h32   = (float*) (ws + 0);          // 49152*64*4  = 12582912
  __bf16* hbf   = (__bf16*)(ws + 12582912);   // 49152*64*2  =  6291456
  float*  proj  = (float*) (ws + 18874368);   // 49152*160*4 = 31457280
  __bf16* hcat  = (__bf16*)(ws + 50331648);   // 49152*128*2 = 12582912
  float*  meta  = (float*) (ws + 62914560);   // 4096*64*4   =  1048576
  __bf16* WtG   = (__bf16*)(ws + 63963136);   // 64*1792*2   =   229376
  __bf16* WptG  = (__bf16*)(ws + 64192512);   // 160*64*2    =    20480
  __bf16* WmtG  = (__bf16*)(ws + 64212992);   // 64*128*2    =    16384
  __bf16* WotG  = (__bf16*)(ws + 64229376);   // 240*64*2    =    30720
  float*  bpcat = (float*) (ws + 64260096);   // 160*4

  const int prep_elems = 64 * K_PAD + NPROJ * 64 + 64 * 128 + NC_PAD * 64 + NPROJ;
  k_prep<<<(prep_elems + 255) / 256, 256, 0, stream>>>(
      W_in, Wp_f, Wp_b, W_merge, W_out, bp_f, bp_b, WtG, WptG, WmtG, WotG,
      bpcat);
  k_meta<<<(B_SZ * 64) / 256, 256, 0, stream>>>(site_ids, hours, site_emb,
                                                hour_emb, W_meta, b_meta, meta);
  k_in<<<M_TOT / 32, 256, 0, stream>>>(emb, scores, b_in, ln_in_g, ln_in_b,
                                       WtG, meta, pos_enc, h32, hbf);
  k_proj<<<M_TOT / 32, 256, 0, stream>>>(hbf, WptG, bpcat, proj);
  k_scan<<<B_SZ, 128, 0, stream>>>(proj, h32, A_log_f, A_log_b, D_f, D_b,
                                   hcat);
  k_out<<<M_TOT / 32, 256, 0, stream>>>(hcat, h32, WmtG, WotG, b_merge, ln_g,
                                        ln_b, b_out, (float*)d_out);
}
